// MultiHeadAttention_39453569581101
// MI455X (gfx1250) — matrix operand order repaired and checked by reading, not compiled, not run on hardware
//
#include <hip/hip_runtime.h>

#define D_MODEL   1024
#define NUM_HEADS 16
#define DEPTH     64
#define BATCH     2
#define SEQ       2048
#define M_TOK     (BATCH * SEQ)   // 4096 token rows

typedef __attribute__((ext_vector_type(16))) __bf16         v16bf;
typedef __attribute__((ext_vector_type(8)))  float          v8f;
typedef __attribute__((ext_vector_type(4)))  float          v4f;
typedef __attribute__((ext_vector_type(8)))  unsigned short v8us;

// float -> bf16 with round-to-nearest-even
__device__ __forceinline__ unsigned short f2bf(float f) {
    unsigned int u = __float_as_uint(f);
    u += 0x7FFFu + ((u >> 16) & 1u);
    return (unsigned short)(u >> 16);
}

__device__ __forceinline__ v8f wmma_acc(v16bf a, v16bf b, v8f c) {
    // (neg_a, A, neg_b, B, c_mod, C, reuse_a, reuse_b)
    return __builtin_amdgcn_wmma_f32_16x16x32_bf16(false, a, false, b,
                                                   (short)0, c, false, false);
}

// A-operand fragment: 16x32 (MxK) bf16, row-major source [*, ld].
// lane L: row = row0 + (L&15); K elems = k0 + (L>>4)*8 + {0..7} and +16..
__device__ __forceinline__ v16bf load_a_bf(const unsigned short* base, int ld,
                                           int row0, int k0) {
    int lane = threadIdx.x & 31;
    const unsigned short* p =
        base + (size_t)(row0 + (lane & 15)) * ld + k0 + ((lane >> 4) << 3);
    union { v16bf v; v8us h[2]; } u;
    u.h[0] = *(const v8us*)p;
    u.h[1] = *(const v8us*)(p + 16);
    return u.v;
}

// Same A layout, f32 source -> bf16 on the fly; 4x 128-bit loads per fragment.
__device__ __forceinline__ v16bf load_a_f32(const float* base, int ld,
                                            int row0, int k0) {
    int lane = threadIdx.x & 31;
    const float* p =
        base + (size_t)(row0 + (lane & 15)) * ld + k0 + ((lane >> 4) << 3);
    v4f f0 = *(const v4f*)(p);
    v4f f1 = *(const v4f*)(p + 4);
    v4f f2 = *(const v4f*)(p + 16);
    v4f f3 = *(const v4f*)(p + 20);
    union { v16bf v; unsigned short s[16]; } u;
#pragma unroll
    for (int i = 0; i < 4; ++i) {
        u.s[i]      = f2bf(f0[i]);
        u.s[4 + i]  = f2bf(f1[i]);
        u.s[8 + i]  = f2bf(f2[i]);
        u.s[12 + i] = f2bf(f3[i]);
    }
    return u.v;
}

// B-operand fragment: 32x16 (KxN) bf16, sourced from a row-major [N, ld]
// buffer (i.e. B^T): each lane reads two runs of 8 K values of its column.
// lane L: col = col0 + (L&15); K elems = k0 + (L>>4)*8 + {0..7} and +16..
__device__ __forceinline__ v16bf load_b_bf(const unsigned short* base, int ld,
                                           int col0, int k0) {
    int lane = threadIdx.x & 31;
    const unsigned short* p =
        base + (size_t)(col0 + (lane & 15)) * ld + k0 + ((lane >> 4) << 3);
    union { v16bf v; v8us h[2]; } u;
    u.h[0] = *(const v8us*)p;
    u.h[1] = *(const v8us*)(p + 16);
    return u.v;
}

// ---------------------------------------------------------------- conversion
// 8 elements / thread: 2x b128 loads, 1x b128 store. n8 = n / 8.
__global__ __launch_bounds__(256) void f32_to_bf16_kernel(
    const float* __restrict__ in, unsigned short* __restrict__ out, size_t n8) {
    size_t i = (size_t)blockIdx.x * blockDim.x + threadIdx.x;
    if (i >= n8) return;
    const v4f* p = (const v4f*)(in + i * 8);
    v4f f0 = p[0];
    v4f f1 = p[1];
    v8us o;
#pragma unroll
    for (int j = 0; j < 4; ++j) {
        o[j]     = f2bf(f0[j]);
        o[4 + j] = f2bf(f1[j]);
    }
    *(v8us*)(out + i * 8) = o;
}

// ------------------------------------------------------------- projection GEMM
// Y = X @ W^T + bias.  X:[M,K] bf16 row-major, W:[N,K] bf16 row-major.
// OUT_MODE 0: bf16 head-split  [B,H,S,DEPTH]
// OUT_MODE 1: bf16 head-split transposed [B,H,DEPTH,S]   (for V)
// OUT_MODE 2: f32 plain [M,N]                            (final output)
// Block: 256 threads = 8 waves; wave tile 32x64; block tile 64x256.
template <int OUT_MODE>
__global__ __launch_bounds__(256) void proj_gemm(
    const unsigned short* __restrict__ X, const unsigned short* __restrict__ W,
    const float* __restrict__ bias, void* __restrict__ out,
    int M, int K, int N) {
    int w    = threadIdx.x >> 5;
    int lane = threadIdx.x & 31;
    int row0 = blockIdx.x * 64  + (w & 1) * 32;
    int col0 = blockIdx.y * 256 + (w >> 1) * 64;

    v8f acc[2][4] = {};
    for (int k = 0; k < K; k += 32) {
        v16bf a0 = load_a_bf(X, K, row0,      k);
        v16bf a1 = load_a_bf(X, K, row0 + 16, k);
#pragma unroll
        for (int nt = 0; nt < 4; ++nt) {
            v16bf b = load_b_bf(W, K, col0 + nt * 16, k);
            acc[0][nt] = wmma_acc(a0, b, acc[0][nt]);
            acc[1][nt] = wmma_acc(a1, b, acc[1][nt]);
        }
    }

    int mBase = row0 + 8 * (lane >> 4);
    int nB    = lane & 15;
#pragma unroll
    for (int mt = 0; mt < 2; ++mt)
#pragma unroll
        for (int nt = 0; nt < 4; ++nt)
#pragma unroll
            for (int r = 0; r < 8; ++r) {
                int m = mBase + mt * 16 + r;
                int n = col0 + nt * 16 + nB;
                float val = acc[mt][nt][r] + bias[n];
                if (OUT_MODE == 2) {
                    ((float*)out)[(size_t)m * N + n] = val;
                } else {
                    int bi = m >> 11;          // m / SEQ
                    int s  = m & (SEQ - 1);
                    int h  = n >> 6;           // n / DEPTH
                    int d  = n & (DEPTH - 1);
                    size_t idx;
                    if (OUT_MODE == 0)
                        idx = (((size_t)(bi * NUM_HEADS + h) * SEQ) + s) * DEPTH + d;
                    else
                        idx = (((size_t)(bi * NUM_HEADS + h) * DEPTH) + d) * SEQ + s;
                    ((unsigned short*)out)[idx] = f2bf(val);
                }
            }
}

// -------------------------------------------------------------- logits GEMM
// logits[b,h,q,kc] = (Qh . Kh) / 8 + mask[b,0,q,kc] * 1e-9  -> f32 weights buf
__global__ __launch_bounds__(256) void logits_gemm(
    const unsigned short* __restrict__ Qh, const unsigned short* __restrict__ Kh,
    const float* __restrict__ mask, float* __restrict__ Wt) {
    int w    = threadIdx.x >> 5;
    int lane = threadIdx.x & 31;
    int bh   = blockIdx.z;
    int bi   = bh >> 4;
    int q0   = blockIdx.x * 64  + (w & 1) * 32;
    int c0   = blockIdx.y * 128 + (w >> 1) * 32;

    const unsigned short* Qb = Qh + (size_t)bh * SEQ * DEPTH;
    const unsigned short* Kb = Kh + (size_t)bh * SEQ * DEPTH;

    v8f acc[2][2] = {};
#pragma unroll
    for (int k = 0; k < DEPTH; k += 32) {
        v16bf a0  = load_a_bf(Qb, DEPTH, q0,      k);
        v16bf a1  = load_a_bf(Qb, DEPTH, q0 + 16, k);
        v16bf kb0 = load_b_bf(Kb, DEPTH, c0,      k);
        v16bf kb1 = load_b_bf(Kb, DEPTH, c0 + 16, k);
        acc[0][0] = wmma_acc(a0, kb0, acc[0][0]);
        acc[0][1] = wmma_acc(a0, kb1, acc[0][1]);
        acc[1][0] = wmma_acc(a1, kb0, acc[1][0]);
        acc[1][1] = wmma_acc(a1, kb1, acc[1][1]);
    }

    const float* mb = mask + (size_t)bi * SEQ * SEQ;
    float*       Wb = Wt   + (size_t)bh * SEQ * SEQ;
    int mBase = q0 + 8 * (lane >> 4);
    int nB    = lane & 15;
#pragma unroll
    for (int mt = 0; mt < 2; ++mt)
#pragma unroll
        for (int nt = 0; nt < 2; ++nt)
#pragma unroll
            for (int r = 0; r < 8; ++r) {
                int qv = mBase + mt * 16 + r;
                int nv = c0 + nt * 16 + nB;
                size_t idx = (size_t)qv * SEQ + nv;
                Wb[idx] = acc[mt][nt][r] * 0.125f + mb[idx] * 1e-9f;
            }
}

// ------------------------------------------------------------------- softmax
// In-place softmax over rows of SEQ floats. One block per row.
__global__ __launch_bounds__(256) void softmax_rows(float* __restrict__ Wt) {
    __shared__ float red[256];
    float* p = Wt + (size_t)blockIdx.x * SEQ;
    int t = threadIdx.x;

    float vals[8];
    float m = -3.402823e38f;
#pragma unroll
    for (int i = 0; i < 8; ++i) { vals[i] = p[t + i * 256]; m = fmaxf(m, vals[i]); }
    red[t] = m;
    __syncthreads();
    for (int off = 128; off > 0; off >>= 1) {
        if (t < off) red[t] = fmaxf(red[t], red[t + off]);
        __syncthreads();
    }
    m = red[0];
    __syncthreads();

    float sum = 0.f;
#pragma unroll
    for (int i = 0; i < 8; ++i) { vals[i] = __expf(vals[i] - m); sum += vals[i]; }
    red[t] = sum;
    __syncthreads();
    for (int off = 128; off > 0; off >>= 1) {
        if (t < off) red[t] += red[t + off];
        __syncthreads();
    }
    float inv = 1.0f / red[0];
#pragma unroll
    for (int i = 0; i < 8; ++i) p[t + i * 256] = vals[i] * inv;
}

// --------------------------------------------------------------- P @ V GEMM
// attn[b,h,q,d] = sum_k P[b,h,q,k] * V[b,h,k,d];  P f32 (converted on the fly),
// Vt is [B,H,DEPTH,SEQ] bf16 so B-fragments load contiguously.
// Wave tile: 16 q-rows x full 64 d-cols -> each A fragment feeds 4 WMMAs.
// Block: 8 waves = 128 q rows. Output merged [B,S,H*DEPTH] bf16.
__global__ __launch_bounds__(256) void attnv_gemm(
    const float* __restrict__ Wt, const unsigned short* __restrict__ Vt,
    unsigned short* __restrict__ Am) {
    int w    = threadIdx.x >> 5;
    int lane = threadIdx.x & 31;
    int bh   = blockIdx.y;
    int bi   = bh >> 4, h = bh & 15;
    int q0   = blockIdx.x * 128 + w * 16;

    const float*          Pb = Wt + (size_t)bh * SEQ * SEQ;
    const unsigned short* Vb = Vt + (size_t)bh * DEPTH * SEQ;

    v8f acc[4] = {};
    for (int k = 0; k < SEQ; k += 32) {
        v16bf a = load_a_f32(Pb, SEQ, q0, k);
#pragma unroll
        for (int nt = 0; nt < 4; ++nt) {
            v16bf b = load_b_bf(Vb, SEQ, nt * 16, k);
            acc[nt] = wmma_acc(a, b, acc[nt]);
        }
    }

    int qBase = q0 + 8 * (lane >> 4);
    int nB    = lane & 15;
#pragma unroll
    for (int r = 0; r < 8; ++r) {
        int qq = qBase + r;
        size_t rowoff = ((size_t)bi * SEQ + qq) * D_MODEL + h * DEPTH;
#pragma unroll
        for (int nt = 0; nt < 4; ++nt)
            Am[rowoff + nt * 16 + nB] = f2bf(acc[nt][r]);
    }
}

// ------------------------------------------------------------------- launch
extern "C" void kernel_launch(void* const* d_in, const int* in_sizes, int n_in,
                              void* d_out, int out_size, void* d_ws, size_t ws_size,
                              hipStream_t stream) {
    const float* q    = (const float*)d_in[0];
    const float* k    = (const float*)d_in[1];
    const float* v    = (const float*)d_in[2];
    const float* mask = (const float*)d_in[3];
    const float* Wq   = (const float*)d_in[4];
    const float* bq   = (const float*)d_in[5];
    const float* Wk   = (const float*)d_in[6];
    const float* bk   = (const float*)d_in[7];
    const float* Wv   = (const float*)d_in[8];
    const float* bv   = (const float*)d_in[9];
    const float* Wo   = (const float*)d_in[10];
    const float* bo   = (const float*)d_in[11];

    float* out     = (float*)d_out;                       // [B,S,D_MODEL]
    float* weights = out + (size_t)M_TOK * D_MODEL;       // [B,H,S,S]

    // workspace carve-up (bf16 elements); ~56 MB total
    const size_t TOKD = (size_t)M_TOK * D_MODEL;          // 4096*1024
    const size_t WSZ  = (size_t)D_MODEL * D_MODEL;
    unsigned short* ws  = (unsigned short*)d_ws;
    size_t off = 0;
    unsigned short* qbf = ws + off; off += TOKD;
    unsigned short* kbf = ws + off; off += TOKD;
    unsigned short* vbf = ws + off; off += TOKD;
    unsigned short* wqb = ws + off; off += WSZ;
    unsigned short* wkb = ws + off; off += WSZ;
    unsigned short* wvb = ws + off; off += WSZ;
    unsigned short* wob = ws + off; off += WSZ;
    unsigned short* qh  = ws + off; off += TOKD;          // [B,H,S,DEPTH]
    unsigned short* kh  = ws + off; off += TOKD;          // [B,H,S,DEPTH]
    unsigned short* vt  = ws + off; off += TOKD;          // [B,H,DEPTH,S]
    unsigned short* am  = qbf;   // reuse: q bf16 dead after the Q projection

    // 1. f32 -> bf16 staging (8 elems / thread, vectorized)
    {
        unsigned int bt = (unsigned int)((TOKD / 8 + 255) / 256);
        unsigned int bw = (unsigned int)((WSZ  / 8 + 255) / 256);
        f32_to_bf16_kernel<<<bt, 256, 0, stream>>>(q,  qbf, TOKD / 8);
        f32_to_bf16_kernel<<<bt, 256, 0, stream>>>(k,  kbf, TOKD / 8);
        f32_to_bf16_kernel<<<bt, 256, 0, stream>>>(v,  vbf, TOKD / 8);
        f32_to_bf16_kernel<<<bw, 256, 0, stream>>>(Wq, wqb, WSZ / 8);
        f32_to_bf16_kernel<<<bw, 256, 0, stream>>>(Wk, wkb, WSZ / 8);
        f32_to_bf16_kernel<<<bw, 256, 0, stream>>>(Wv, wvb, WSZ / 8);
        f32_to_bf16_kernel<<<bw, 256, 0, stream>>>(Wo, wob, WSZ / 8);
    }

    // 2. Q/K/V projections (WMMA)
    dim3 gProj(M_TOK / 64, D_MODEL / 256);                // 64 x 4
    proj_gemm<0><<<gProj, 256, 0, stream>>>(qbf, wqb, bq, qh, M_TOK, D_MODEL, D_MODEL);
    proj_gemm<0><<<gProj, 256, 0, stream>>>(kbf, wkb, bk, kh, M_TOK, D_MODEL, D_MODEL);
    proj_gemm<1><<<gProj, 256, 0, stream>>>(vbf, wvb, bv, vt, M_TOK, D_MODEL, D_MODEL);

    // 3. logits = QK^T/sqrt(d) + mask*1e-9  (WMMA, f32 out to d_out weights)
    dim3 gLog(SEQ / 64, SEQ / 128, BATCH * NUM_HEADS);    // 32 x 16 x 32
    logits_gemm<<<gLog, 256, 0, stream>>>(qh, kh, mask, weights);

    // 4. softmax in place over last dim
    softmax_rows<<<dim3(BATCH * NUM_HEADS * SEQ), 256, 0, stream>>>(weights);

    // 5. attn = P @ V (WMMA, A converted f32->bf16 with 128-bit loads)
    attnv_gemm<<<dim3(SEQ / 128, BATCH * NUM_HEADS), 256, 0, stream>>>(weights, vt, am);

    // 6. out = attn @ Wo^T + bo (WMMA, f32 out)
    proj_gemm<2><<<gProj, 256, 0, stream>>>(am, wob, bo, out, M_TOK, D_MODEL, D_MODEL);
}
